// _AttentionLayer_5652176961576
// MI455X (gfx1250) — matrix operand order repaired and checked by reading, not compiled, not run on hardware
//
#include <hip/hip_runtime.h>
#include <hip/hip_bf16.h>
#include <stdint.h>

// ---------------------------------------------------------------------------
// MI455X / gfx1250, wave32. All matmuls via v_wmma_f32_16x16x32_bf16.
// ---------------------------------------------------------------------------

typedef __attribute__((ext_vector_type(16))) __bf16 v16bf;
typedef __attribute__((ext_vector_type(8)))  float  v8f;

union FragU {
  v16bf v;
  uint4 q[2];
};

#if defined(__has_builtin)
#if __has_builtin(__builtin_amdgcn_global_load_async_to_lds_b128)
#define USE_ASYNC_LDS 1
#endif
#endif
#ifndef USE_ASYNC_LDS
#define USE_ASYNC_LDS 0
#endif

// raw 16-byte int vector type as expected by the async-LDS builtin
typedef int async_i4 __attribute__((vector_size(16)));
#define AS_GLOBAL __attribute__((address_space(1)))
#define AS_LDS __attribute__((address_space(3)))

// 16-byte global -> LDS copy; async (ASYNCcnt-tracked, no VGPR round trip)
// when the toolchain exposes the gfx1250 builtin, plain copy otherwise.
__device__ __forceinline__ void copy16_g2l(void* lds_dst, const void* gsrc) {
#if USE_ASYNC_LDS
  __builtin_amdgcn_global_load_async_to_lds_b128(
      (AS_GLOBAL async_i4*)(gsrc),
      (AS_LDS async_i4*)(lds_dst), 0, 0);
#else
  *(uint4*)lds_dst = *(const uint4*)gsrc;
#endif
}

__device__ __forceinline__ void async_fence() {
#if USE_ASYNC_LDS
  asm volatile("s_wait_asynccnt 0" ::: "memory");
#endif
}

__device__ __forceinline__ v8f wmma_bf16(v16bf a, v16bf b, v8f c) {
  return __builtin_amdgcn_wmma_f32_16x16x32_bf16(
      /*neg_a=*/false, a, /*neg_b=*/false, b,
      /*c_mod=*/(short)0, c, /*reuse_a=*/false, /*reuse_b=*/false);
}

// A-matrix fragment (16x32 bf16) from LDS tile, row-major [M][K].
// ISA layout: lanes 0-15 hold M rows; VGPR0-3 = K0..7, VGPR4-7 = K16..23;
// lanes 16-31: VGPR0-3 = K8..15, VGPR4-7 = K24..31.
__device__ __forceinline__ v16bf ldsA_frag(const __bf16* tile, int row0,
                                           int stride_elems, int koff) {
  const int lane = threadIdx.x & 31;
  const int half = lane >> 4;
  const uint4* p4 =
      (const uint4*)(tile + (size_t)(row0 + (lane & 15)) * stride_elems + koff);
  FragU f;
  f.q[0] = p4[half];      // dwords 4*half   .. 4*half+3
  f.q[1] = p4[2 + half];  // dwords 8+4*half .. 8+4*half+3
  return f.v;
}

// B-matrix fragment (32x16 bf16) from LDS tile staged TRANSPOSED as [N][K].
// ISA layout: lanes 0-15 = N cols holding K0..7,K16..23; lanes 16-31 hold K8..15,K24..31.
__device__ __forceinline__ v16bf ldsB_frag(const __bf16* tile, int col0,
                                           int stride_elems, int koff) {
  const int lane = threadIdx.x & 31;
  const int half = lane >> 4;
  const uint4* p4 =
      (const uint4*)(tile + (size_t)(col0 + (lane & 15)) * stride_elems + koff);
  FragU f;
  f.q[0] = p4[half];
  f.q[1] = p4[2 + half];
  return f.v;
}

// pack two bf16 into one dword (for b32 transpose staging)
__device__ __forceinline__ uint32_t pack2(__bf16 lo, __bf16 hi) {
  union {
    __bf16 h[2];
    uint32_t u;
  } p;
  p.h[0] = lo;
  p.h[1] = hi;
  return p.u;
}

// ---------------------------------------------------------------------------
// fp32 -> bf16 cast, 8 elems/thread (2x float4 load -> 1x uint4 store)
// ---------------------------------------------------------------------------
__global__ void k_cast_bf16(const float4* __restrict__ src,
                            uint4* __restrict__ dst, int n8) {
  for (int i = blockIdx.x * blockDim.x + threadIdx.x; i < n8;
       i += gridDim.x * blockDim.x) {
    const float4 a = src[2 * i];
    const float4 b = src[2 * i + 1];
    union {
      __bf16 h[8];
      uint4 q;
    } u;
    u.h[0] = (__bf16)a.x; u.h[1] = (__bf16)a.y;
    u.h[2] = (__bf16)a.z; u.h[3] = (__bf16)a.w;
    u.h[4] = (__bf16)b.x; u.h[5] = (__bf16)b.y;
    u.h[6] = (__bf16)b.z; u.h[7] = (__bf16)b.w;
    dst[i] = u.q;
  }
}

// ---------------------------------------------------------------------------
// Generic bf16 GEMM:  C[M,N] = A[M,K] @ B[K,N] + bias   (fp32 accumulate)
// Block tile 128x128, BK=32, 256 threads = 8 waves in 2(M) x 4(N) grid,
// wave tile 64x32 = 4x2 WMMA 16x16 tiles. Output mode is a template param
// so the epilogue is branch-free.
// ---------------------------------------------------------------------------
#define GEMM_BM 128
#define GEMM_BN 128
#define GEMM_BK 32
#define GEMM_LDST 48  // LDS row stride in elems (32 + 16 pad, 96B = 16B mult)

template <bool WRITE_F32, bool WRITE_BF16>
__global__ __launch_bounds__(256) void k_gemm_bf16(
    const __bf16* __restrict__ A, const __bf16* __restrict__ B,
    const float* __restrict__ bias, float* __restrict__ Cf,
    __bf16* __restrict__ Cb, int M, int N, int K) {
  __shared__ __bf16 As[GEMM_BM * GEMM_LDST];
  __shared__ __bf16 Bs[GEMM_BN * GEMM_LDST];

  const int tid = threadIdx.x;
  const int lane = tid & 31;
  const int wave = tid >> 5;
  const int wm = wave & 1;   // 0..1  -> 64-row slab
  const int wn = wave >> 1;  // 0..3  -> 32-col slab
  const int bm = blockIdx.x * GEMM_BM;
  const int bn = blockIdx.y * GEMM_BN;

  // per-thread staging coords
  const int a_r = tid >> 1;            // A: row 0..127
  const int a_c = (tid & 1) * 16;      // A: col 0 / 16
  const int b_kp = (tid >> 4) * 2;     // B: k-pair base 0,2,..,30
  const int b_c = (tid & 15) * 8;      // B: col seg 0..120

  v8f acc[4][2];
  const v8f vzero = {0.f, 0.f, 0.f, 0.f, 0.f, 0.f, 0.f, 0.f};
#pragma unroll
  for (int i = 0; i < 4; ++i)
#pragma unroll
    for (int j = 0; j < 2; ++j) acc[i][j] = vzero;

  for (int k0 = 0; k0 < K; k0 += GEMM_BK) {
    __syncthreads();
    // --- stage A tile [128 x 32] row-major (async path if available) ----
    {
      const __bf16* src = A + (size_t)(bm + a_r) * K + k0 + a_c;
      __bf16* dst = As + a_r * GEMM_LDST + a_c;
      copy16_g2l(dst, src);
      copy16_g2l(dst + 8, src + 8);
    }
    // --- stage B tile [32 x 128] transposed into Bs[N][K], b32-packed ---
    {
      const __bf16* s0 = B + (size_t)(k0 + b_kp) * N + bn + b_c;
      const __bf16* s1 = s0 + N;
      union {
        uint4 q;
        __bf16 h[8];
      } r0, r1;
      r0.q = *(const uint4*)s0;
      r1.q = *(const uint4*)s1;
#pragma unroll
      for (int i = 0; i < 8; ++i)
        *(uint32_t*)(Bs + (b_c + i) * GEMM_LDST + b_kp) = pack2(r0.h[i], r1.h[i]);
    }
    // prefetch next K tiles while this one is consumed
    if (k0 + GEMM_BK < K) {
      __builtin_prefetch(A + (size_t)(bm + a_r) * K + k0 + GEMM_BK + a_c, 0, 1);
      __builtin_prefetch(B + (size_t)(k0 + GEMM_BK + b_kp) * N + bn + b_c, 0, 1);
    }
    async_fence();
    __syncthreads();

    v16bf afr[4], bfr[2];
#pragma unroll
    for (int i = 0; i < 4; ++i)
      afr[i] = ldsA_frag(As, wm * 64 + i * 16, GEMM_LDST, 0);
#pragma unroll
    for (int j = 0; j < 2; ++j)
      bfr[j] = ldsB_frag(Bs, wn * 32 + j * 16, GEMM_LDST, 0);
#pragma unroll
    for (int i = 0; i < 4; ++i)
#pragma unroll
      for (int j = 0; j < 2; ++j) acc[i][j] = wmma_bf16(afr[i], bfr[j], acc[i][j]);
  }

  // --- epilogue: bias add, compile-time-selected stores -----------------
  const int half = lane >> 4;
  const int nlo = lane & 15;
#pragma unroll
  for (int j = 0; j < 2; ++j) {
    const int col = bn + wn * 32 + j * 16 + nlo;
    const float bv = bias[col];
#pragma unroll
    for (int i = 0; i < 4; ++i) {
      const size_t rbase = (size_t)(bm + wm * 64 + i * 16 + half * 8) * N + col;
#pragma unroll
      for (int r = 0; r < 8; ++r) {
        const float v = acc[i][j][r] + bv;
        if constexpr (WRITE_F32) Cf[rbase + (size_t)r * N] = v;
        if constexpr (WRITE_BF16) Cb[rbase + (size_t)r * N] = (__bf16)v;
      }
    }
  }
}

// ---------------------------------------------------------------------------
// Flash attention (unscaled, unmasked):  O = softmax(Q Kᵀ) V
// Q,K,V bf16 [B,S,D], D=256.  Block = 256 threads (8 waves) handles 64 query
// rows; KV streamed in 64-row tiles with fp32 online softmax.
// ---------------------------------------------------------------------------
#define FA_Q 64
#define FA_KV 64
#define FA_D 256
#define FA_QSTR (FA_D + 8)   // 264 elems (528B, 16B-mult)
#define FA_VSTR (FA_KV + 8)  // 72 elems (144B, 16B-mult)
#define FA_SSTR (FA_KV + 4)  // 68 f32

// LDS carve (bytes): Qs 33792 | Ks 33792 | Vts 36864 | Ps 9216 | Ss 17408
//                    | rowm/rowl/rowscale 768  -> total 131840
#define FA_SMEM_BYTES 131840

__global__ __launch_bounds__(256) void k_flash_attn(
    const __bf16* __restrict__ Q, const __bf16* __restrict__ Kg,
    const __bf16* __restrict__ V, float* __restrict__ O, int S) {
  extern __shared__ char smem_raw[];
  __bf16* Qs = (__bf16*)smem_raw;               // [FA_Q][FA_QSTR]
  __bf16* Ks = Qs + FA_Q * FA_QSTR;             // [FA_KV][FA_QSTR] (== Bᵀ stage)
  __bf16* Vts = Ks + FA_KV * FA_QSTR;           // [FA_D][FA_VSTR] (V transposed)
  __bf16* Ps = Vts + FA_D * FA_VSTR;            // [FA_Q][FA_VSTR] (P, A-layout)
  float* Ss = (float*)(Ps + FA_Q * FA_VSTR);    // [FA_Q][FA_SSTR]
  float* rowm = Ss + FA_Q * FA_SSTR;
  float* rowl = rowm + FA_Q;
  float* rowscale = rowl + FA_Q;

  const int tid = threadIdx.x;
  const int lane = tid & 31;
  const int wave = tid >> 5;
  const int half = lane >> 4;
  const int nlo = lane & 15;
  const int wm = wave & 1;   // S-tile / O-tile row slab (32 rows)
  const int wn = wave >> 1;  // S-tile: 16-col slab; O-tile: 64-col slab
  const int qbase = blockIdx.x * FA_Q;
  const size_t boff = (size_t)blockIdx.y * S * FA_D;

  // staging coords
  const int qk_r = tid >> 2;           // Q/K row 0..63
  const int qk_c = (tid & 3) * 64;     // Q/K col seg
  const int v_kvp = (tid >> 3) * 2;    // V kv-pair base 0,2,..,62
  const int v_c = (tid & 7) * 32;      // V d seg 0..224

  // ---- load Q tile once (async path if available) ----------------------
  {
    const __bf16* src = Q + boff + (size_t)(qbase + qk_r) * FA_D + qk_c;
    __bf16* dst = Qs + qk_r * FA_QSTR + qk_c;
#pragma unroll
    for (int i = 0; i < 8; ++i) copy16_g2l(dst + 8 * i, src + 8 * i);
  }
  if (tid < FA_Q) {
    rowm[tid] = -1.0e30f;
    rowl[tid] = 0.f;
  }

  const v8f vzero = {0.f, 0.f, 0.f, 0.f, 0.f, 0.f, 0.f, 0.f};
  v8f oacc[2][4];
#pragma unroll
  for (int i = 0; i < 2; ++i)
#pragma unroll
    for (int j = 0; j < 4; ++j) oacc[i][j] = vzero;

  for (int t = 0; t < S; t += FA_KV) {
    __syncthreads();  // prior-iter LDS readers done; Q stage visible (iter 0)
    // ---- stage K tile (row-major == Bᵀ stage for Q@Kᵀ) -----------------
    {
      const __bf16* src = Kg + boff + (size_t)(t + qk_r) * FA_D + qk_c;
      __bf16* dst = Ks + qk_r * FA_QSTR + qk_c;
#pragma unroll
      for (int i = 0; i < 8; ++i) copy16_g2l(dst + 8 * i, src + 8 * i);
    }
    // ---- stage V tile transposed Vts[d][kv], b32-packed kv pairs -------
    {
      const __bf16* s0 = V + boff + (size_t)(t + v_kvp) * FA_D + v_c;
      const __bf16* s1 = s0 + FA_D;
      union {
        uint4 q[4];
        __bf16 h[32];
      } r0, r1;
#pragma unroll
      for (int j = 0; j < 4; ++j) {
        r0.q[j] = ((const uint4*)s0)[j];
        r1.q[j] = ((const uint4*)s1)[j];
      }
#pragma unroll
      for (int i = 0; i < 32; ++i)
        *(uint32_t*)(Vts + (v_c + i) * FA_VSTR + v_kvp) =
            pack2(r0.h[i], r1.h[i]);
    }
    async_fence();
    __syncthreads();

    // ---- S = Q Kᵀ : wave computes 32x16 ( 2 WMMAs per K-step ) ---------
    v8f sacc[2] = {vzero, vzero};
#pragma unroll
    for (int kk = 0; kk < FA_D; kk += 32) {
      const v16bf bk = ldsB_frag(Ks, wn * 16, FA_QSTR, kk);
      sacc[0] = wmma_bf16(ldsA_frag(Qs, wm * 32, FA_QSTR, kk), bk, sacc[0]);
      sacc[1] = wmma_bf16(ldsA_frag(Qs, wm * 32 + 16, FA_QSTR, kk), bk, sacc[1]);
    }
#pragma unroll
    for (int i = 0; i < 2; ++i)
#pragma unroll
      for (int r = 0; r < 8; ++r) {
        const int row = wm * 32 + i * 16 + half * 8 + r;
        Ss[row * FA_SSTR + wn * 16 + nlo] = sacc[i][r];
      }
    __syncthreads();

    // ---- online softmax over this 64-wide slab (one row per thread) ----
    if (tid < FA_Q) {
      const float* srow = Ss + tid * FA_SSTR;
      float mnew = rowm[tid];
#pragma unroll 8
      for (int c = 0; c < FA_KV; ++c) mnew = fmaxf(mnew, srow[c]);
      const float sc = __expf(rowm[tid] - mnew);
      float l = rowl[tid] * sc;
      __bf16* prow = Ps + tid * FA_VSTR;
#pragma unroll 8
      for (int c = 0; c < FA_KV; ++c) {
        const float p = __expf(srow[c] - mnew);
        l += p;
        prow[c] = (__bf16)p;
      }
      rowm[tid] = mnew;
      rowl[tid] = l;
      rowscale[tid] = sc;
    }
    __syncthreads();

    // ---- rescale running O, then O += P V ------------------------------
#pragma unroll
    for (int i = 0; i < 2; ++i)
#pragma unroll
      for (int r = 0; r < 8; ++r) {
        const float sc = rowscale[wm * 32 + i * 16 + half * 8 + r];
#pragma unroll
        for (int j = 0; j < 4; ++j) oacc[i][j][r] *= sc;
      }
#pragma unroll
    for (int kk = 0; kk < FA_KV; kk += 32) {
      const v16bf ap0 = ldsA_frag(Ps, wm * 32, FA_VSTR, kk);
      const v16bf ap1 = ldsA_frag(Ps, wm * 32 + 16, FA_VSTR, kk);
#pragma unroll
      for (int j = 0; j < 4; ++j) {
        const v16bf bv = ldsB_frag(Vts, wn * 64 + j * 16, FA_VSTR, kk);
        oacc[0][j] = wmma_bf16(ap0, bv, oacc[0][j]);
        oacc[1][j] = wmma_bf16(ap1, bv, oacc[1][j]);
      }
    }
  }

  // ---- epilogue: O /= l ------------------------------------------------
#pragma unroll
  for (int i = 0; i < 2; ++i)
#pragma unroll
    for (int r = 0; r < 8; ++r) {
      const int row = wm * 32 + i * 16 + half * 8 + r;
      const float inv = 1.0f / rowl[row];
#pragma unroll
      for (int j = 0; j < 4; ++j) {
        const int col = wn * 64 + j * 16 + nlo;
        O[boff + (size_t)(qbase + row) * FA_D + col] = oacc[i][j][r] * inv;
      }
    }
}

// ---------------------------------------------------------------------------
// LayerNorm helpers (wave32 shuffles + LDS)
// ---------------------------------------------------------------------------
__device__ __forceinline__ float blockReduceSum(float v, float* sh) {
  __syncthreads();
  const int lane = threadIdx.x & 31;
  const int w = threadIdx.x >> 5;
#pragma unroll
  for (int o = 16; o; o >>= 1) v += __shfl_xor(v, o, 32);
  if (lane == 0) sh[w] = v;
  __syncthreads();
  float r = 0.f;
  const int nw = (int)(blockDim.x >> 5);
  if ((int)threadIdx.x < nw) r = sh[threadIdx.x];
  if (w == 0) {
#pragma unroll
    for (int o = 16; o; o >>= 1) r += __shfl_xor(r, o, 32);
    if (lane == 0) sh[0] = r;
  }
  __syncthreads();
  return sh[0];
}

// t1 = LN(x + attn) -> bf16     (C == blockDim.x == 256)
__global__ __launch_bounds__(256) void k_ln_residual_bf16(
    const float* __restrict__ x, const float* __restrict__ a,
    const float* __restrict__ g, const float* __restrict__ b,
    __bf16* __restrict__ out, int C) {
  __shared__ float sh[32];
  const size_t base = (size_t)blockIdx.x * C;
  const float v = x[base + threadIdx.x] + a[base + threadIdx.x];
  const float mean = blockReduceSum(v, sh) / (float)C;
  const float d = v - mean;
  const float var = blockReduceSum(d * d, sh) / (float)C;
  const float rs = rsqrtf(var + 1e-5f);
  out[base + threadIdx.x] = (__bf16)(d * rs * g[threadIdx.x] + b[threadIdx.x]);
}

// h = leaky_relu(LN(h)) in place on bf16 buffer, C = 2048 (8 per thread)
__global__ __launch_bounds__(256) void k_ln_leaky_inplace(
    __bf16* __restrict__ h, const float* __restrict__ g,
    const float* __restrict__ b, int C) {
  __shared__ float sh[32];
  __bf16* p = h + (size_t)blockIdx.x * C;
  float v[8];
  float s = 0.f;
#pragma unroll
  for (int i = 0; i < 8; ++i) {
    v[i] = (float)p[threadIdx.x + i * 256];
    s += v[i];
  }
  const float mean = blockReduceSum(s, sh) / (float)C;
  float var = 0.f;
#pragma unroll
  for (int i = 0; i < 8; ++i) {
    const float d = v[i] - mean;
    var += d * d;
  }
  var = blockReduceSum(var, sh) / (float)C;
  const float rs = rsqrtf(var + 1e-5f);
#pragma unroll
  for (int i = 0; i < 8; ++i) {
    const int c = threadIdx.x + i * 256;
    float y = (v[i] - mean) * rs * g[c] + b[c];
    y = (y > 0.f) ? y : 0.01f * y;
    p[c] = (__bf16)y;
  }
}

// out = LN(leaky_relu(logits + e)) -> fp32   (C == 256)
__global__ __launch_bounds__(256) void k_ln_final(
    const float* __restrict__ logits, const float* __restrict__ e,
    const float* __restrict__ g, const float* __restrict__ b,
    float* __restrict__ out, int C) {
  __shared__ float sh[32];
  const size_t base = (size_t)blockIdx.x * C;
  float v = logits[base + threadIdx.x] + e[base + threadIdx.x];
  v = (v > 0.f) ? v : 0.01f * v;
  const float mean = blockReduceSum(v, sh) / (float)C;
  const float d = v - mean;
  const float var = blockReduceSum(d * d, sh) / (float)C;
  const float rs = rsqrtf(var + 1e-5f);
  out[base + threadIdx.x] = d * rs * g[threadIdx.x] + b[threadIdx.x];
}

// ---------------------------------------------------------------------------
// Host orchestration
// ---------------------------------------------------------------------------
extern "C" void kernel_launch(void* const* d_in, const int* in_sizes, int n_in,
                              void* d_out, int out_size, void* d_ws,
                              size_t ws_size, hipStream_t stream) {
  constexpr int Bsz = 4, S = 4096, E = 256, H = 2048;
  constexpr int M = Bsz * S;  // 16384 rows

  const float* x = (const float*)d_in[0];
  const float* wq = (const float*)d_in[1];
  const float* bq = (const float*)d_in[2];
  const float* wk = (const float*)d_in[3];
  const float* bk = (const float*)d_in[4];
  const float* wv = (const float*)d_in[5];
  const float* bv = (const float*)d_in[6];
  const float* w0 = (const float*)d_in[7];
  const float* b0 = (const float*)d_in[8];
  const float* w1 = (const float*)d_in[9];
  const float* b1 = (const float*)d_in[10];
  const float* g_mlp = (const float*)d_in[11];
  const float* b_mlp = (const float*)d_in[12];
  const float* w2 = (const float*)d_in[13];
  const float* b2 = (const float*)d_in[14];
  const float* g1 = (const float*)d_in[15];
  const float* bln1 = (const float*)d_in[16];
  const float* g2 = (const float*)d_in[17];
  const float* bln2 = (const float*)d_in[18];
  float* out = (float*)d_out;

  // ---- workspace carve -------------------------------------------------
  char* ws = (char*)d_ws;
  auto carve = [&](size_t bytes) {
    char* p = ws;
    ws += (bytes + 255) & ~(size_t)255;
    return p;
  };
  __bf16* xb = (__bf16*)carve((size_t)M * E * 2);
  __bf16* wqb = (__bf16*)carve((size_t)E * E * 2);
  __bf16* wkb = (__bf16*)carve((size_t)E * E * 2);
  __bf16* wvb = (__bf16*)carve((size_t)E * E * 2);
  __bf16* w0b = (__bf16*)carve((size_t)E * E * 2);
  __bf16* w1b = (__bf16*)carve((size_t)E * H * 2);
  __bf16* w2b = (__bf16*)carve((size_t)H * E * 2);
  __bf16* qb = (__bf16*)carve((size_t)M * E * 2);
  __bf16* kb = (__bf16*)carve((size_t)M * E * 2);
  __bf16* vb = (__bf16*)carve((size_t)M * E * 2);
  float* attn = (float*)carve((size_t)M * E * 4);
  __bf16* t1b = (__bf16*)carve((size_t)M * E * 2);
  float* e_f = (float*)carve((size_t)M * E * 4);
  __bf16* eb = (__bf16*)carve((size_t)M * E * 2);
  __bf16* hb = (__bf16*)carve((size_t)M * H * 2);
  float* logits = (float*)carve((size_t)M * E * 4);

  // ---- 1) fp32 -> bf16 casts ------------------------------------------
  k_cast_bf16<<<512, 256, 0, stream>>>((const float4*)x, (uint4*)xb,
                                       M * E / 8);
  k_cast_bf16<<<32, 256, 0, stream>>>((const float4*)wq, (uint4*)wqb,
                                      E * E / 8);
  k_cast_bf16<<<32, 256, 0, stream>>>((const float4*)wk, (uint4*)wkb,
                                      E * E / 8);
  k_cast_bf16<<<32, 256, 0, stream>>>((const float4*)wv, (uint4*)wvb,
                                      E * E / 8);
  k_cast_bf16<<<32, 256, 0, stream>>>((const float4*)w0, (uint4*)w0b,
                                      E * E / 8);
  k_cast_bf16<<<256, 256, 0, stream>>>((const float4*)w1, (uint4*)w1b,
                                       E * H / 8);
  k_cast_bf16<<<256, 256, 0, stream>>>((const float4*)w2, (uint4*)w2b,
                                       H * E / 8);

  // ---- 2) QKV projections (bf16 out) ----------------------------------
  {
    dim3 g(M / GEMM_BM, E / GEMM_BN);
    k_gemm_bf16<false, true>
        <<<g, 256, 0, stream>>>(xb, wqb, bq, nullptr, qb, M, E, E);
    k_gemm_bf16<false, true>
        <<<g, 256, 0, stream>>>(xb, wkb, bk, nullptr, kb, M, E, E);
    k_gemm_bf16<false, true>
        <<<g, 256, 0, stream>>>(xb, wvb, bv, nullptr, vb, M, E, E);
  }

  // ---- 3) flash attention ---------------------------------------------
  {
    dim3 g(S / FA_Q, Bsz);
    k_flash_attn<<<g, 256, FA_SMEM_BYTES, stream>>>(qb, kb, vb, attn, S);
  }

  // ---- 4) t1 = LN(x + attn) -> bf16 -----------------------------------
  k_ln_residual_bf16<<<M, 256, 0, stream>>>(x, attn, g1, bln1, t1b, E);

  // ---- 5) e = t1 @ w0 + b0  (fp32 + bf16 out) -------------------------
  {
    dim3 g(M / GEMM_BM, E / GEMM_BN);
    k_gemm_bf16<true, true>
        <<<g, 256, 0, stream>>>(t1b, w0b, b0, e_f, eb, M, E, E);
  }

  // ---- 6) h = e @ w1 + b1 (bf16 out), then h = leaky(LN(h)) in place --
  {
    dim3 g(M / GEMM_BM, H / GEMM_BN);
    k_gemm_bf16<false, true>
        <<<g, 256, 0, stream>>>(eb, w1b, b1, nullptr, hb, M, H, E);
  }
  k_ln_leaky_inplace<<<M, 256, 0, stream>>>(hb, g_mlp, b_mlp, H);

  // ---- 7) logits = h @ w2 + b2 (fp32 out) -----------------------------
  {
    dim3 g(M / GEMM_BM, E / GEMM_BN);
    k_gemm_bf16<true, false>
        <<<g, 256, 0, stream>>>(hb, w2b, b2, logits, nullptr, M, E, H);
  }

  // ---- 8) out = LN(leaky(logits + e)) ---------------------------------
  k_ln_final<<<M, 256, 0, stream>>>(logits, e_f, g2, bln2, out, E);
}
